// AttentionBlock_29265907155317
// MI455X (gfx1250) — matrix operand order repaired and checked by reading, not compiled, not run on hardware
//
#include <hip/hip_runtime.h>
#include <hip/hip_bf16.h>

typedef __attribute__((ext_vector_type(16))) _Float16 v16h;
typedef __attribute__((ext_vector_type(8)))  _Float16 v8h;
typedef __attribute__((ext_vector_type(8)))  float    v8f;

#define WMMA_F32_F16(a, b, c) \
  __builtin_amdgcn_wmma_f32_16x16x32_f16(false, (a), false, (b), (short)0, (c), false, false)

__device__ static inline v16h pack16(v8h lo, v8h hi) {
  v16h r;
#pragma unroll
  for (int i = 0; i < 8; ++i) { r[i] = lo[i]; r[i + 8] = hi[i]; }
  return r;
}

__device__ static inline v8f vzero8f() {
  v8f r;
#pragma unroll
  for (int i = 0; i < 8; ++i) r[i] = 0.0f;
  return r;
}

// ---- 16-lane (DPP-row) reductions: pure VALU, no LDS traffic ----
template <int CTRL>
__device__ static inline float dppmov(float v) {
  return __int_as_float(
      __builtin_amdgcn_update_dpp(0, __float_as_int(v), CTRL, 0xf, 0xf, true));
}
__device__ static inline float red16_max(float v) {
  v = fmaxf(v, dppmov<0xB1>(v));   // quad_perm(1,0,3,2)  : xor 1
  v = fmaxf(v, dppmov<0x4E>(v));   // quad_perm(2,3,0,1)  : xor 2
  v = fmaxf(v, dppmov<0x141>(v));  // row_half_mirror     : crosses quads in 8
  v = fmaxf(v, dppmov<0x140>(v));  // row_mirror          : crosses 8s in 16
  return v;
}
__device__ static inline float red16_add(float v) {
  v += dppmov<0xB1>(v);
  v += dppmov<0x4E>(v);
  v += dppmov<0x141>(v);
  v += dppmov<0x140>(v);
  return v;
}

// ---------------- BitLinear weight quantization ----------------

__global__ __launch_bounds__(256) void absmean_partial(const float* __restrict__ W, int n,
                                                       float* __restrict__ part) {
  __shared__ float red[256];
  float s = 0.0f;
  for (int i = blockIdx.x * blockDim.x + threadIdx.x; i < n; i += gridDim.x * blockDim.x)
    s += fabsf(W[i]);
  red[threadIdx.x] = s;
  __syncthreads();
  for (int st = 128; st > 0; st >>= 1) {
    if (threadIdx.x < st) red[threadIdx.x] += red[threadIdx.x + st];
    __syncthreads();
  }
  if (threadIdx.x == 0) part[blockIdx.x] = red[0];
}

__global__ __launch_bounds__(256) void absmean_final(const float* __restrict__ part, int nb,
                                                     float n, float* __restrict__ scale) {
  __shared__ float red[256];
  float s = 0.0f;
  for (int i = threadIdx.x; i < nb; i += 256) s += part[i];
  red[threadIdx.x] = s;
  __syncthreads();
  for (int st = 128; st > 0; st >>= 1) {
    if (threadIdx.x < st) red[threadIdx.x] += red[threadIdx.x + st];
    __syncthreads();
  }
  if (threadIdx.x == 0) *scale = fmaxf(red[0] / n, 1e-5f);
}

__global__ __launch_bounds__(256) void quantize_w(const float* __restrict__ W,
                                                  const float* __restrict__ scale_p, int n,
                                                  _Float16* __restrict__ out) {
  float s = *scale_p;
  for (int i = blockIdx.x * blockDim.x + threadIdx.x; i < n; i += gridDim.x * blockDim.x) {
    float q = rintf(W[i] / s);         // round-half-even, matches jnp.round
    q = fminf(1.0f, fmaxf(-1.0f, q));
    out[i] = (_Float16)(q * s);
  }
}

// ---------------- RMSNorm (fp32 in -> f16 out) ----------------

__global__ __launch_bounds__(256) void rmsnorm_k(const float* __restrict__ x,
                                                 const float* __restrict__ gamma,
                                                 _Float16* __restrict__ out, int D) {
  __shared__ float red[256];
  int row = blockIdx.x;
  const float* xr = x + (size_t)row * D;
  float s = 0.0f;
  for (int i = threadIdx.x; i < D; i += 256) { float v = xr[i]; s += v * v; }
  red[threadIdx.x] = s;
  __syncthreads();
  for (int st = 128; st > 0; st >>= 1) {
    if (threadIdx.x < st) red[threadIdx.x] += red[threadIdx.x + st];
    __syncthreads();
  }
  float rinv = rsqrtf(red[0] / (float)D + 1e-6f);
  for (int i = threadIdx.x; i < D; i += 256)
    out[(size_t)row * D + i] = (_Float16)(xr[i] * rinv * gamma[i]);
}

// ---------------- WMMA GEMM: out[M,N] = A[M,K] @ Bw[N,K]^T (+resid) ----------------
// 256 threads = 8 wave32 (4x2), block tile 128x64, wave tile 32x32, K-slab 32.
// Double-buffered LDS: next slab's global loads issue before this slab's WMMAs;
// one s_barrier per slab.

template <bool HALF_OUT>
__global__ __launch_bounds__(256) void gemm_wmma(const _Float16* __restrict__ A,
                                                 const _Float16* __restrict__ Bw,
                                                 const float* __restrict__ resid,
                                                 void* __restrict__ outp,
                                                 int M, int N, int K) {
  __shared__ _Float16 As[2][128 * 40];  // 80B row pitch: 16B-aligned, bank-spread
  __shared__ _Float16 Bs[2][64 * 40];
  const int tid = threadIdx.x;
  const int wid = tid >> 5, lane = tid & 31, g = lane >> 4, ln = lane & 15;
  const int wm = wid >> 1, wn = wid & 1;
  const int m0 = blockIdx.y * 128, n0 = blockIdx.x * 64;
  const int r0 = tid >> 2, c0 = (tid & 3) * 8;  // A chunks: rows r0, r0+64; B chunk: row r0

  v8f acc[2][2];
#pragma unroll
  for (int mi = 0; mi < 2; ++mi)
#pragma unroll
    for (int ni = 0; ni < 2; ++ni) acc[mi][ni] = vzero8f();

  // prologue: stage slab 0
  {
    v8h a0 = *(const v8h*)&A[(size_t)(m0 + r0) * K + c0];
    v8h a1 = *(const v8h*)&A[(size_t)(m0 + r0 + 64) * K + c0];
    v8h b0 = *(const v8h*)&Bw[(size_t)(n0 + r0) * K + c0];
    *(v8h*)&As[0][r0 * 40 + c0] = a0;
    *(v8h*)&As[0][(r0 + 64) * 40 + c0] = a1;
    *(v8h*)&Bs[0][r0 * 40 + c0] = b0;
  }
  __syncthreads();

  const int nk = K >> 5;
  for (int i = 0; i < nk; ++i) {
    const int cur = i & 1, nxt = cur ^ 1;
    const bool more = (i + 1 < nk);
    v8h na0, na1, nb0;
    if (more) {  // issue next slab's global loads early (latency hides under WMMAs)
      const int k1 = (i + 1) << 5;
      na0 = *(const v8h*)&A[(size_t)(m0 + r0) * K + k1 + c0];
      na1 = *(const v8h*)&A[(size_t)(m0 + r0 + 64) * K + k1 + c0];
      nb0 = *(const v8h*)&Bw[(size_t)(n0 + r0) * K + k1 + c0];
    }

    v16h af[2], bf[2];
#pragma unroll
    for (int mi = 0; mi < 2; ++mi) {
      const _Float16* p = &As[cur][(wm * 32 + mi * 16 + ln) * 40];
      af[mi] = pack16(*(const v8h*)(p + g * 8), *(const v8h*)(p + 16 + g * 8));
    }
#pragma unroll
    for (int ni = 0; ni < 2; ++ni) {
      const _Float16* p = &Bs[cur][(wn * 32 + ni * 16 + ln) * 40 + g * 8];
      bf[ni] = pack16(*(const v8h*)p, *(const v8h*)(p + 16));
    }
#pragma unroll
    for (int mi = 0; mi < 2; ++mi)
#pragma unroll
      for (int ni = 0; ni < 2; ++ni)
        acc[mi][ni] = WMMA_F32_F16(af[mi], bf[ni], acc[mi][ni]);

    if (more) {  // safe: previous barrier guaranteed all reads of 'nxt' finished
      *(v8h*)&As[nxt][r0 * 40 + c0] = na0;
      *(v8h*)&As[nxt][(r0 + 64) * 40 + c0] = na1;
      *(v8h*)&Bs[nxt][r0 * 40 + c0] = nb0;
    }
    __syncthreads();
  }

#pragma unroll
  for (int mi = 0; mi < 2; ++mi)
#pragma unroll
    for (int ni = 0; ni < 2; ++ni)
#pragma unroll
      for (int r = 0; r < 8; ++r) {
        int m = m0 + wm * 32 + mi * 16 + r + 8 * g;  // C-frag: row = r + 8g, col = ln
        int n = n0 + wn * 32 + ni * 16 + ln;
        size_t off = (size_t)m * N + n;
        float v = acc[mi][ni][r];
        if (resid) v += resid[off];
        if (HALF_OUT) ((_Float16*)outp)[off] = (_Float16)v;
        else          ((float*)outp)[off] = v;
      }
}

// ---------------- V transpose: qkv[B*T,3072] -> Vt[b][h][d][t] ----------------

__global__ __launch_bounds__(256) void transpose_v(const _Float16* __restrict__ qkvb,
                                                   _Float16* __restrict__ vt) {
  int i = blockIdx.x * 256 + threadIdx.x;  // total 2*16*64*2048 = 4M
  int t = i & 2047;
  int r = i >> 11;
  int d = r & 63; r >>= 6;
  int h = r & 15;
  int b = r >> 4;
  vt[i] = qkvb[((size_t)(b * 2048 + t)) * 3072 + 2048 + h * 64 + d];
}

// ---------------- Flash attention (causal, online softmax, all-WMMA) ----------------
// grid = (B*H, T/128); 8 waves/block, wave owns 16 queries; 32-key blocks.
// Softmax row stats via DPP-row reductions (no LDS); P re-layout via per-wave LDS.

__global__ __launch_bounds__(256) void flash_attn_kernel(const _Float16* __restrict__ qkvb,
                                                         const _Float16* __restrict__ vt,
                                                         _Float16* __restrict__ attn) {
  const int T = 2048, H = 16, LD = 3072, Dh = 64;
  int b = blockIdx.x >> 4, h = blockIdx.x & 15;
  int wid = threadIdx.x >> 5, lane = threadIdx.x & 31;
  int g = lane >> 4, ln = lane & 15;
  int qtile = blockIdx.y * 128 + wid * 16;

  const _Float16* qrow = qkvb + ((size_t)(b * T) + qtile + ln) * LD + h * Dh;
  const _Float16* kbp  = qkvb + (size_t)(b * T) * LD + 1024 + h * Dh;
  const _Float16* vtp  = vt + (size_t)(b * H + h) * Dh * T;  // [d][t]

  __shared__ _Float16 pl[8][16 * 32];  // per-wave P staging (no block barriers)
  _Float16* myp = pl[wid];

  v16h qf[2];
#pragma unroll
  for (int s = 0; s < 2; ++s) {
    int d0 = 32 * s;
    qf[s] = pack16(*(const v8h*)(qrow + d0 + g * 8),
                   *(const v8h*)(qrow + d0 + 16 + g * 8));
  }

  v8f o[4];
#pragma unroll
  for (int f = 0; f < 4; ++f) o[f] = vzero8f();
  float m_run[8], l_run[8];
#pragma unroll
  for (int r = 0; r < 8; ++r) { m_run[r] = -__builtin_inff(); l_run[r] = 0.0f; }

  const float sm_scale = 0.125f;  // Dh^-0.5

  for (int kb = 0; kb < qtile + 16; kb += 32) {
    // ---- S = Q @ K^T over two 16-key n-tiles ----
    v8f sfr[2];
#pragma unroll
    for (int j = 0; j < 2; ++j) {
      v8f a = vzero8f();
#pragma unroll
      for (int s = 0; s < 2; ++s) {
        const _Float16* kr = kbp + (size_t)(kb + 16 * j + ln) * LD + 32 * s + g * 8;
        a = WMMA_F32_F16(qf[s], pack16(*(const v8h*)kr, *(const v8h*)(kr + 16)), a);
      }
      sfr[j] = a;
    }
#pragma unroll
    for (int r = 0; r < 8; ++r) { sfr[0][r] *= sm_scale; sfr[1][r] *= sm_scale; }

    if (kb + 31 > qtile) {  // causal mask on diagonal blocks
#pragma unroll
      for (int r = 0; r < 8; ++r) {
        int q = qtile + r + 8 * g;
        sfr[0][r] = (kb + ln      > q) ? -__builtin_inff() : sfr[0][r];
        sfr[1][r] = (kb + 16 + ln > q) ? -__builtin_inff() : sfr[1][r];
      }
    }

    // ---- online softmax (row = r + 8g lives in one 16-lane DPP row) ----
    float alpha[8];
#pragma unroll
    for (int r = 0; r < 8; ++r) {
      float v = red16_max(fmaxf(sfr[0][r], sfr[1][r]));
      float mn = fmaxf(m_run[r], v);
      alpha[r] = __expf(m_run[r] - mn);
      m_run[r] = mn;
    }
#pragma unroll
    for (int r = 0; r < 8; ++r) {
      float p0 = __expf(sfr[0][r] - m_run[r]);
      float p1 = __expf(sfr[1][r] - m_run[r]);
      myp[(r + 8 * g) * 32 + ln]      = (_Float16)p0;
      myp[(r + 8 * g) * 32 + 16 + ln] = (_Float16)p1;
      l_run[r] = l_run[r] * alpha[r] + red16_add(p0 + p1);
    }
#pragma unroll
    for (int f = 0; f < 4; ++f)
#pragma unroll
      for (int r = 0; r < 8; ++r) o[f][r] *= alpha[r];

    asm volatile("s_wait_dscnt 0" ::: "memory");  // P stores visible before frag reads

    // ---- P re-layout (LDS round-trip) and O += P @ V ----
    v16h pf = pack16(*(const v8h*)&myp[ln * 32 + g * 8],
                     *(const v8h*)&myp[ln * 32 + 16 + g * 8]);
#pragma unroll
    for (int f = 0; f < 4; ++f) {
      const _Float16* vr = vtp + (size_t)(16 * f + ln) * T + kb + g * 8;
      o[f] = WMMA_F32_F16(pf, pack16(*(const v8h*)vr, *(const v8h*)(vr + 16)), o[f]);
    }
  }

#pragma unroll
  for (int f = 0; f < 4; ++f)
#pragma unroll
    for (int r = 0; r < 8; ++r) {
      int q = qtile + r + 8 * g;
      attn[((size_t)(b * T) + q) * 1024 + h * 64 + 16 * f + ln] =
          (_Float16)(o[f][r] / l_run[r]);
    }
}

// ---------------- SiLU(gate) * up (in-place into gate buffer) ----------------

__global__ __launch_bounds__(256) void silu_mul(const _Float16* __restrict__ gate,
                                                const _Float16* __restrict__ up,
                                                _Float16* __restrict__ act, int n) {
  for (int i = blockIdx.x * blockDim.x + threadIdx.x; i < n; i += gridDim.x * blockDim.x) {
    float gv = (float)gate[i];
    float s = gv / (1.0f + __expf(-gv));
    act[i] = (_Float16)(s * (float)up[i]);
  }
}

// ---------------- host-side launch ----------------

static void* bump(char** p, size_t bytes) {
  void* r = (void*)*p;
  *p += (bytes + 255) & ~(size_t)255;
  return r;
}

extern "C" void kernel_launch(void* const* d_in, const int* in_sizes, int n_in,
                              void* d_out, int out_size, void* d_ws, size_t ws_size,
                              hipStream_t stream) {
  const float* x      = (const float*)d_in[0];
  const float* ln1_w  = (const float*)d_in[1];
  const float* qkv_w  = (const float*)d_in[2];
  const float* out_w  = (const float*)d_in[3];
  const float* ln2_w  = (const float*)d_in[4];
  const float* gate_w = (const float*)d_in[5];
  const float* up_w   = (const float*)d_in[6];
  const float* down_w = (const float*)d_in[7];

  const int B = 2, T = 2048, D = 1024, H = 16, Dh = 64, F = 4096;
  const int M = B * T, NQKV = 3 * H * Dh;

  char* p = (char*)d_ws;
  float*    scales   = (float*)bump(&p, 8 * sizeof(float));
  float*    partials = (float*)bump(&p, 5 * 512 * sizeof(float));
  _Float16* wq_qkv   = (_Float16*)bump(&p, (size_t)NQKV * D * 2);
  _Float16* wq_out   = (_Float16*)bump(&p, (size_t)D * D * 2);
  _Float16* wq_gate  = (_Float16*)bump(&p, (size_t)F * D * 2);
  _Float16* wq_up    = (_Float16*)bump(&p, (size_t)F * D * 2);
  _Float16* wq_down  = (_Float16*)bump(&p, (size_t)D * F * 2);
  _Float16* h1       = (_Float16*)bump(&p, (size_t)M * D * 2);
  _Float16* qkvb     = (_Float16*)bump(&p, (size_t)M * NQKV * 2);
  _Float16* vtb      = (_Float16*)bump(&p, (size_t)B * H * Dh * T * 2);
  _Float16* attn     = (_Float16*)bump(&p, (size_t)M * D * 2);
  float*    x1       = (float*)bump(&p, (size_t)M * D * 4);
  _Float16* h2       = (_Float16*)bump(&p, (size_t)M * D * 2);
  _Float16* gateb    = (_Float16*)bump(&p, (size_t)M * F * 2);
  _Float16* upb      = (_Float16*)bump(&p, (size_t)M * F * 2);

  struct WQ { const float* w; int n; _Float16* out; };
  WQ wq[5] = {{qkv_w, NQKV * D, wq_qkv}, {out_w, D * D, wq_out},
              {gate_w, F * D, wq_gate},  {up_w, F * D, wq_up},
              {down_w, D * F, wq_down}};
  for (int i = 0; i < 5; ++i) {
    absmean_partial<<<512, 256, 0, stream>>>(wq[i].w, wq[i].n, partials + i * 512);
    absmean_final<<<1, 256, 0, stream>>>(partials + i * 512, 512, (float)wq[i].n, scales + i);
    quantize_w<<<2048, 256, 0, stream>>>(wq[i].w, scales + i, wq[i].n, wq[i].out);
  }

  // h1 = rmsnorm(x, ln1); qkv = h1 @ Wqkv^T
  rmsnorm_k<<<M, 256, 0, stream>>>(x, ln1_w, h1, D);
  gemm_wmma<true><<<dim3(NQKV / 64, M / 128), 256, 0, stream>>>(h1, wq_qkv, nullptr, qkvb, M, NQKV, D);

  // attention
  transpose_v<<<(B * H * Dh * T) / 256, 256, 0, stream>>>(qkvb, vtb);
  flash_attn_kernel<<<dim3(B * H, T / 128), 256, 0, stream>>>(qkvb, vtb, attn);

  // x1 = x + attn @ Wout^T
  gemm_wmma<false><<<dim3(D / 64, M / 128), 256, 0, stream>>>(attn, wq_out, x, x1, M, D, D);

  // FFN: x1 + down( silu(gate(h2)) * up(h2) )
  rmsnorm_k<<<M, 256, 0, stream>>>(x1, ln2_w, h2, D);
  gemm_wmma<true><<<dim3(F / 64, M / 128), 256, 0, stream>>>(h2, wq_gate, nullptr, gateb, M, F, D);
  gemm_wmma<true><<<dim3(F / 64, M / 128), 256, 0, stream>>>(h2, wq_up, nullptr, upb, M, F, D);
  silu_mul<<<4096, 256, 0, stream>>>(gateb, upb, gateb, M * F);
  gemm_wmma<false><<<dim3(D / 64, M / 128), 256, 0, stream>>>(gateb, wq_down, x1, (float*)d_out, M, D, F);
}
